// OffsetAttention_6665789243939
// MI455X (gfx1250) — matrix operand order repaired and checked by reading, not compiled, not run on hardware
//
#include <hip/hip_runtime.h>
#include <hip/hip_bf16.h>

typedef _Float16 f16;
typedef __attribute__((ext_vector_type(16))) _Float16 v16h;
typedef __attribute__((ext_vector_type(8)))  _Float16 v8h;
typedef __attribute__((ext_vector_type(8)))  float    v8f;

#define B_  4
#define C_  256
#define CQ_ 64
#define N_  4096

// ---------------------------------------------------------------------------
// WMMA fragment helpers (CDNA5 16x16x32 f16, wave32).
// A 16x32 (MxK): lane m = L&15; lanes 0-15 carry K=0..15, lanes 16-31 carry
// K=16..31 (matching B) => two contiguous 8-half (16B) loads per lane.
__device__ inline v16h load_a_frag(const f16* __restrict__ p, int ld) {
  const int lane = threadIdx.x & 31;
  const f16* r = p + (lane & 15) * ld + ((lane >> 4) << 4);
  v8h lo = *(const v8h*)(r);
  v8h hi = *(const v8h*)(r + 8);
  v16h a;
#pragma unroll
  for (int i = 0; i < 8; ++i) { a[i] = lo[i]; a[i + 8] = hi[i]; }
  return a;
}

// B 32x16 (KxN): lane holds one column n = L&15; lanes 0-15 carry K=0..15,
// lanes 16-31 carry K=16..31. Source stored K-contiguous per column => one
// 16-half (32B) load per lane.
__device__ inline v16h load_b_frag(const f16* __restrict__ p, int ld) {
  const int lane = threadIdx.x & 31;
  return *(const v16h*)(p + (lane & 15) * ld + ((lane >> 4) << 4));
}

__device__ inline v8f wmma_f16(v16h a, v16h b, v8f c) {
  return __builtin_amdgcn_wmma_f32_16x16x32_f16(false, a, false, b, (short)0, c,
                                                false, false);
}

// ---------------------------------------------------------------------------
// K0: zero column-sum accumulator, convert w_trans to f16 (GEMM A operand).
__global__ void kInit(const float* __restrict__ wt, f16* __restrict__ wt_h,
                      float* __restrict__ colsum) {
  const int i = blockIdx.x * 256 + threadIdx.x;      // 65536 threads
  wt_h[i] = (f16)wt[i];
  if (i < B_ * N_) colsum[i] = 0.f;
}

// K1a: P[b][n][q] = sum_c w_qk[q][c] * x[b][c][n]  (f16 out, q contiguous).
__global__ void kProjP(const float* __restrict__ x, const float* __restrict__ wqk,
                       f16* __restrict__ P) {
  const int idx = blockIdx.x * 256 + threadIdx.x;    // ((b*CQ + q)*N + n)
  const int n = idx & (N_ - 1);
  const int q = (idx >> 12) & (CQ_ - 1);
  const int b = idx >> 18;
  const float* xb = x + (size_t)b * C_ * N_ + n;
  const float* w  = wqk + q * C_;                    // uniform in block -> SMEM
  float acc = 0.f;
#pragma unroll 4
  for (int c = 0; c < C_; ++c) acc += w[c] * xb[(size_t)c * N_];
  P[((size_t)(b * N_ + n)) * CQ_ + q] = (f16)acc;
}

// K1b: V[b][o][n] = sum_c w_v[o][c] * x[b][c][n] + b_v[o]  (f16, n contiguous).
__global__ void kProjV(const float* __restrict__ x, const float* __restrict__ wv,
                       const float* __restrict__ bv, f16* __restrict__ V) {
  const int idx = blockIdx.x * 256 + threadIdx.x;    // ((b*C + o)*N + n)
  const int n = idx & (N_ - 1);
  const int o = (idx >> 12) & (C_ - 1);
  const int b = idx >> 20;
  const float* xb = x + (size_t)b * C_ * N_ + n;
  const float* w  = wv + o * C_;                     // uniform in block -> SMEM
  float acc = bv[o];
#pragma unroll 4
  for (int c = 0; c < C_; ++c) acc += w[c] * xb[(size_t)c * N_];
  V[idx] = (f16)acc;
}

// ---------------------------------------------------------------------------
// K2: one workgroup = 16 score rows x 4096 cols (256KB LDS, legal on CDNA5's
// 320KB/WGP). scores = P Pt via WMMA -> row softmax in LDS -> write attnT
// (f16, transposed so the next GEMM gets contiguous B fragments), and
// atomically accumulate column sums for the later L1 renorm.
__global__ __launch_bounds__(128) void kScoresSoftmax(
    const f16* __restrict__ P, f16* __restrict__ attnT, float* __restrict__ colsum) {
  extern __shared__ float sc[];                      // 16 * N_ floats = 256 KB
  __shared__ float red[16][8];
  __shared__ float rowmax[16];
  __shared__ float rowinv[16];

  const int b    = blockIdx.x >> 8;                  // 256 row-blocks per batch
  const int row0 = (blockIdx.x & 255) << 4;
  const int tid  = threadIdx.x;
  const int wv   = tid >> 5;
  const int lane = tid & 31;

  const f16* Pb = P + (size_t)b * N_ * CQ_;

  // A fragments: 16 rows x K=64 of P, kept in registers for all 64 tiles.
  const f16* arow = Pb + (size_t)row0 * CQ_;
  const v16h a0 = load_a_frag(arow, CQ_);
  const v16h a1 = load_a_frag(arow + 32, CQ_);

  const int col0 = wv << 10;                         // each wave: 1024 columns
  for (int t = 0; t < 64; ++t) {
    const int colt = col0 + (t << 4);
    const f16* bp = Pb + (size_t)colt * CQ_;         // B(k=q, col=m) = P[m][q]
    v8f c = {};
    c = wmma_f16(a0, load_b_frag(bp, CQ_), c);
    c = wmma_f16(a1, load_b_frag(bp + 32, CQ_), c);
    const int cc  = colt + (lane & 15);
    const int mrb = (lane >> 4) << 3;
#pragma unroll
    for (int r = 0; r < 8; ++r) sc[(mrb + r) * N_ + cc] = c[r];
  }
  __syncthreads();

  // Row max: 8 threads per row, 512 cols each.
  const int rr = tid >> 3, ss = tid & 7;
  {
    const float* rp = sc + rr * N_ + (ss << 9);
    float mx = rp[0];
    for (int i = 1; i < 512; ++i) mx = fmaxf(mx, rp[i]);
    red[rr][ss] = mx;
  }
  __syncthreads();
  if (tid < 16) {
    float mx = red[tid][0];
#pragma unroll
    for (int s = 1; s < 8; ++s) mx = fmaxf(mx, red[tid][s]);
    rowmax[tid] = mx;
  }
  __syncthreads();
  {
    const float mx = rowmax[rr];
    float* rp = sc + rr * N_ + (ss << 9);
    float sum = 0.f;
    for (int i = 0; i < 512; ++i) {
      const float v = __expf(rp[i] - mx);
      rp[i] = v;
      sum += v;
    }
    red[rr][ss] = sum;
  }
  __syncthreads();
  if (tid < 16) {
    float sum = 0.f;
#pragma unroll
    for (int s = 0; s < 8; ++s) sum += red[tid][s];
    rowinv[tid] = 1.f / sum;                         // >= 1, exp(0) term present
  }
  __syncthreads();

  // Column pass: write attn transposed (n contiguous) + accumulate colsum.
  for (int col = tid; col < N_; col += 128) {
    float cs = 0.f;
    v16h outv;
#pragma unroll
    for (int r = 0; r < 16; ++r) {
      const float v = sc[r * N_ + col] * rowinv[r];
      cs += v;
      outv[r] = (f16)v;
    }
    *(v16h*)(attnT + ((size_t)(b * N_ + col)) * N_ + row0) = outv;
    atomicAdd(colsum + b * N_ + col, cs);
  }
}

// ---------------------------------------------------------------------------
// K3: new = V(256xN) @ attn(NxN) via WMMA (K=4096) with 2x2 register blocking
//     (wave computes a 32x32 output block: 4 WMMAs per 4 fragment loads, 2x
//     arithmetic intensity vs 16x16). Fused column renorm and residual:
//     RT[b][m][c] = x[b][c][m] - new[c][m] / (1e-9 + colsum[m]).
__global__ __launch_bounds__(256) void kNewFeatResidual(
    const f16* __restrict__ V, const f16* __restrict__ attnT,
    const float* __restrict__ colsum, const float* __restrict__ x,
    f16* __restrict__ RT) {
  const int g    = blockIdx.x * 8 + (threadIdx.x >> 5); // wave id
  const int lane = threadIdx.x & 31;
  const int b   = g >> 10;                              // 8*128 tiles / batch
  const int mt2 = (g >> 7) & 7;                         // 32-row blocks
  const int nt2 = g & 127;                              // 32-col blocks

  const f16* A0 = V + ((size_t)b * C_ + (mt2 << 5)) * N_;
  const f16* A1 = A0 + (size_t)16 * N_;
  const f16* B0 = attnT + ((size_t)(b * N_ + (nt2 << 5))) * N_;
  const f16* B1 = B0 + (size_t)16 * N_;

  v8f c00 = {}, c01 = {}, c10 = {}, c11 = {};
  for (int k = 0; k < N_; k += 32) {
    __builtin_prefetch(B0 + k + 256, 0, 0);             // global_prefetch_b8
    __builtin_prefetch(B1 + k + 256, 0, 0);
    const v16h a0 = load_a_frag(A0 + k, N_);
    const v16h a1 = load_a_frag(A1 + k, N_);
    const v16h b0 = load_b_frag(B0 + k, N_);
    const v16h b1 = load_b_frag(B1 + k, N_);
    c00 = wmma_f16(a0, b0, c00);
    c01 = wmma_f16(a0, b1, c01);
    c10 = wmma_f16(a1, b0, c10);
    c11 = wmma_f16(a1, b1, c11);
  }

  const int mrb  = (lane >> 4) << 3;
  const int col0 = (nt2 << 5) + (lane & 15);
  const float inv0 = 1.f / (1e-9f + colsum[b * N_ + col0]);
  const float inv1 = 1.f / (1e-9f + colsum[b * N_ + col0 + 16]);
  const v8f acc[2][2] = {{c00, c01}, {c10, c11}};
#pragma unroll
  for (int i = 0; i < 2; ++i) {
#pragma unroll
    for (int j = 0; j < 2; ++j) {
      const int col = col0 + (j << 4);
      const int row = (mt2 << 5) + (i << 4) + mrb;
      const float inv = j ? inv1 : inv0;
      const float* xb = x + ((size_t)b * C_ + row) * N_ + col;
      v8h outv;
#pragma unroll
      for (int r = 0; r < 8; ++r)
        outv[r] = (f16)(xb[(size_t)r * N_] - acc[i][j][r] * inv);
      *(v8h*)(RT + ((size_t)(b * N_ + col)) * C_ + row) = outv;
    }
  }
}

// ---------------------------------------------------------------------------
// K4: y = w_trans(256x256) @ RT' (K=256) + b_trans, BN(inference), ReLU.
//     Same 2x2 register blocking.
__global__ __launch_bounds__(256) void kTransBNRelu(
    const f16* __restrict__ Wt, const f16* __restrict__ RT,
    const float* __restrict__ bt, const float* __restrict__ gamma,
    const float* __restrict__ beta, const float* __restrict__ mean,
    const float* __restrict__ var, float* __restrict__ out) {
  const int g    = blockIdx.x * 8 + (threadIdx.x >> 5);
  const int lane = threadIdx.x & 31;
  const int b   = g >> 10;
  const int mt2 = (g >> 7) & 7;
  const int nt2 = g & 127;

  const f16* A0 = Wt + (mt2 << 5) * C_;
  const f16* A1 = A0 + 16 * C_;
  const f16* B0 = RT + ((size_t)(b * N_ + (nt2 << 5))) * C_;
  const f16* B1 = B0 + 16 * C_;

  v8f c00 = {}, c01 = {}, c10 = {}, c11 = {};
#pragma unroll
  for (int k = 0; k < C_; k += 32) {
    const v16h a0 = load_a_frag(A0 + k, C_);
    const v16h a1 = load_a_frag(A1 + k, C_);
    const v16h b0 = load_b_frag(B0 + k, C_);
    const v16h b1 = load_b_frag(B1 + k, C_);
    c00 = wmma_f16(a0, b0, c00);
    c01 = wmma_f16(a0, b1, c01);
    c10 = wmma_f16(a1, b0, c10);
    c11 = wmma_f16(a1, b1, c11);
  }

  const int mrb  = (lane >> 4) << 3;
  const int col0 = (nt2 << 5) + (lane & 15);
  const v8f acc[2][2] = {{c00, c01}, {c10, c11}};
#pragma unroll
  for (int i = 0; i < 2; ++i) {
#pragma unroll
    for (int j = 0; j < 2; ++j) {
      const int col = col0 + (j << 4);
#pragma unroll
      for (int r = 0; r < 8; ++r) {
        const int row = (mt2 << 5) + (i << 4) + mrb + r;
        float t = acc[i][j][r] + bt[row];
        t = (t - mean[row]) * (gamma[row] * rsqrtf(var[row] + 1e-5f)) + beta[row];
        out[((size_t)b * C_ + row) * N_ + col] = fmaxf(t, 0.f);
      }
    }
  }
}

// ---------------------------------------------------------------------------
extern "C" void kernel_launch(void* const* d_in, const int* in_sizes, int n_in,
                              void* d_out, int out_size, void* d_ws, size_t ws_size,
                              hipStream_t stream) {
  const float* x     = (const float*)d_in[0];
  const float* wqk   = (const float*)d_in[1];
  const float* wv    = (const float*)d_in[2];
  const float* bv    = (const float*)d_in[3];
  const float* wt    = (const float*)d_in[4];
  const float* bt    = (const float*)d_in[5];
  const float* gamma = (const float*)d_in[6];
  const float* beta  = (const float*)d_in[7];
  const float* mean  = (const float*)d_in[8];
  const float* var   = (const float*)d_in[9];
  float* out = (float*)d_out;

  char* ws = (char*)d_ws;
  size_t off = 0;
  f16*   P      = (f16*)(ws + off);   off += (size_t)B_ * N_ * CQ_ * 2;  //   2 MB
  f16*   V      = (f16*)(ws + off);   off += (size_t)B_ * C_ * N_ * 2;   //   8 MB
  f16*   RT     = (f16*)(ws + off);   off += (size_t)B_ * N_ * C_ * 2;   //   8 MB
  f16*   WtH    = (f16*)(ws + off);   off += (size_t)C_ * C_ * 2;        // 128 KB
  float* colsum = (float*)(ws + off); off += (size_t)B_ * N_ * 4;        //  64 KB
  off = (off + 255) & ~(size_t)255;
  f16*   attnT  = (f16*)(ws + off);                                      // 134 MB

  kInit<<<(C_ * C_) / 256, 256, 0, stream>>>(wt, WtH, colsum);
  kProjP<<<(B_ * CQ_ * N_) / 256, 256, 0, stream>>>(x, wqk, P);
  kProjV<<<(B_ * C_ * N_) / 256, 256, 0, stream>>>(x, wv, bv, V);
  kScoresSoftmax<<<B_ * (N_ / 16), 128, 16 * N_ * sizeof(float), stream>>>(
      P, attnT, colsum);
  kNewFeatResidual<<<(B_ * (C_ / 32) * (N_ / 32)) / 8, 256, 0, stream>>>(
      V, attnT, colsum, x, RT);
  kTransBNRelu<<<(B_ * (C_ / 32) * (N_ / 32)) / 8, 256, 0, stream>>>(
      WtH, RT, bt, gamma, beta, mean, var, out);
}
